// DenoisyNet_89790586290204
// MI455X (gfx1250) — matrix operand order repaired and checked by reading, not compiled, not run on hardware
//
#include <hip/hip_runtime.h>
#include <stdint.h>

#define HH 1024
#define WW 1024
#define HWPX (HH * WW)

typedef __attribute__((ext_vector_type(16))) __bf16 v16bf;
typedef __attribute__((ext_vector_type(8)))  float  v8f;
typedef __attribute__((ext_vector_type(4)))  int    v4i;

union V16BF {
    uint4 q[2];
    v16bf v;
};

#if __has_builtin(__builtin_amdgcn_global_load_async_to_lds_b128) && \
    __has_builtin(__builtin_amdgcn_s_wait_asynccnt)
#define HAS_ASYNC 1
#else
#define HAS_ASYNC 0
#endif

static __device__ __forceinline__ void copy16_g2l(unsigned short* lp, const unsigned short* gp) {
#if HAS_ASYNC
    __builtin_amdgcn_global_load_async_to_lds_b128(
        (__attribute__((address_space(1))) v4i*)gp,
        (__attribute__((address_space(3))) v4i*)lp, 0, 0);
#else
    *(uint4*)lp = *(const uint4*)gp;
#endif
}
static __device__ __forceinline__ void async_join() {
#if HAS_ASYNC
    __builtin_amdgcn_s_wait_asynccnt(0);
#endif
    __syncthreads();
}

static __device__ __forceinline__ unsigned short f2bfbits(float f) {
    unsigned int u = __builtin_bit_cast(unsigned int, f);
    unsigned int lsb = (u >> 16) & 1u;
    u += 0x7fffu + lsb;
    return (unsigned short)(u >> 16);
}
static __device__ __forceinline__ float bf2f(unsigned short s) {
    unsigned int u = ((unsigned int)s) << 16;
    return __builtin_bit_cast(float, u);
}

// ---------------------------------------------------------------------------
// Pack OIHW f32 weights into the WMMA B-matrix layout, bf16 pairs per dword.
// dst layout: [g][tap][nt][lane(32)][dword(8)]
// lane l: n = nt*16 + (l&15); dword v: k = (l>>4)*16 + 2v  (k within 32-group)
// weight index: o = n, i = ci0 + g*32 + k   (w is [Cout][CinW][3][3])
// ---------------------------------------------------------------------------
__global__ void prep_pack(const float* __restrict__ w, uint32_t* __restrict__ dst,
                          int CinW, int ci0, int G, int NT, int CoutValid) {
    int idx = blockIdx.x * blockDim.x + threadIdx.x;
    int total = G * 9 * NT * 32 * 8;
    if (idx >= total) return;
    int v = idx & 7;
    int rest = idx >> 3;
    int lane = rest & 31; rest >>= 5;
    int nt = rest % NT;   rest /= NT;
    int t = rest % 9;
    int g = rest / 9;
    int n = nt * 16 + (lane & 15);
    int kbase = ((lane >> 4) * 16) + v * 2;
    uint32_t out = 0;
    for (int p = 0; p < 2; ++p) {
        int i = ci0 + g * 32 + kbase + p;
        float wv = 0.f;
        if (n < CoutValid && i < CinW) wv = w[(n * CinW + i) * 9 + t];
        out |= ((uint32_t)f2bfbits(wv)) << (16 * p);
    }
    dst[idx] = out;
}

// ---------------------------------------------------------------------------
// 3x3 conv, Cin=1 (f32 plane) -> 32ch bf16 HWC, PReLU, optional bf16 skip add.
// ---------------------------------------------------------------------------
__global__ __launch_bounds__(256)
void conv1to32(const float* __restrict__ in, const float* __restrict__ w,
               const float* __restrict__ b, const float* __restrict__ alphap,
               const unsigned short* __restrict__ skip, unsigned short* __restrict__ out) {
    __shared__ float sw[288];
    __shared__ float sb[32];
    int tid = threadIdx.x;
    for (int i = tid; i < 288; i += 256) sw[i] = w[i];
    if (tid < 32) sb[tid] = b[tid];
    __syncthreads();

    int pix = blockIdx.x * 256 + tid;
    int x = pix % WW, y = pix / WW;
    float tap[9];
#pragma unroll
    for (int t = 0; t < 9; ++t) {
        int py = y + t / 3 - 1, px = x + t % 3 - 1;
        tap[t] = (py >= 0 && py < HH && px >= 0 && px < WW) ? in[py * WW + px] : 0.f;
    }
    float al = *alphap;
    unsigned short* orow = out + (size_t)pix * 32;
    const unsigned short* srow = skip ? (skip + (size_t)pix * 32) : nullptr;
#pragma unroll
    for (int o = 0; o < 32; ++o) {
        float acc = sb[o];
#pragma unroll
        for (int t = 0; t < 9; ++t) acc += sw[o * 9 + t] * tap[t];
        acc = (acc >= 0.f) ? acc : al * acc;
        if (srow) acc += bf2f(srow[o]);
        orow[o] = f2bfbits(acc);
    }
}

// ---------------------------------------------------------------------------
// 3x3 conv, Cin = 32*G (bf16 HWC) -> 32ch bf16 HWC, PReLU.  WMMA core with
// async global->LDS staging of the 10x18-pixel halo tile (8x16 output tile,
// one row per wave, 8 waves per block).
// ---------------------------------------------------------------------------
template <int G>
__global__ __launch_bounds__(256)
void conv32_lds(const unsigned short* __restrict__ in0,
                const unsigned short* __restrict__ in1,
                const uint32_t* __restrict__ pw,
                const float* __restrict__ bias,
                const float* __restrict__ alphap,
                unsigned short* __restrict__ out) {
    __shared__ __align__(16) unsigned short tileA[G * 10 * 18 * 32];
    int tid = threadIdx.x;
    int bx = blockIdx.x % (WW / 16);
    int by = blockIdx.x / (WW / 16);
    int x0 = bx * 16, y0 = by * 8;

    const int CHUNKS = G * 10 * 18 * 4; // 16-byte chunks
    bool edge = (bx == 0) | (bx == WW / 16 - 1) | (by == 0) | (by == HH / 8 - 1);
    if (edge) {
        uint4 z = make_uint4(0, 0, 0, 0);
        for (int c = tid; c < CHUNKS; c += 256) ((uint4*)tileA)[c] = z;
        __syncthreads();
    }
    for (int c = tid; c < CHUNKS; c += 256) {
        int qi = c & 3;
        int px_in = (c >> 2) % 18;
        int rest = (c >> 2) / 18;
        int row = rest % 10;
        int g = rest / 10;
        int py = y0 - 1 + row;
        int px = x0 - 1 + px_in;
        if (py >= 0 && py < HH && px >= 0 && px < WW) {
            const unsigned short* gp = (g ? in1 : in0) + ((size_t)py * WW + px) * 32 + qi * 8;
            unsigned short* lp = tileA + ((g * 10 + row) * 18 + px_in) * 32 + qi * 8;
            copy16_g2l(lp, gp);
        }
    }
    async_join();

    int lane = tid & 31, wv = tid >> 5;
    int m = lane & 15, h = lane >> 4;
    int y = y0 + wv;

    v8f acc0 = {0.f, 0.f, 0.f, 0.f, 0.f, 0.f, 0.f, 0.f};
    v8f acc1 = {0.f, 0.f, 0.f, 0.f, 0.f, 0.f, 0.f, 0.f};

#pragma unroll
    for (int g = 0; g < G; ++g) {
#pragma unroll
        for (int t = 0; t < 9; ++t) {
            int row = wv + t / 3;
            int col = m + t % 3;
            const unsigned short* ap = tileA + ((g * 10 + row) * 18 + col) * 32;
            V16BF A;
            A.q[0] = *(const uint4*)(ap + 16 * h);
            A.q[1] = *(const uint4*)(ap + 16 * h + 8);
            const uint4* bp0 = (const uint4*)(pw + ((size_t)(((g * 9 + t) * 2 + 0) * 32 + lane)) * 8);
            const uint4* bp1 = (const uint4*)(pw + ((size_t)(((g * 9 + t) * 2 + 1) * 32 + lane)) * 8);
            V16BF B0, B1;
            B0.q[0] = bp0[0]; B0.q[1] = bp0[1];
            B1.q[0] = bp1[0]; B1.q[1] = bp1[1];
            acc0 = __builtin_amdgcn_wmma_f32_16x16x32_bf16(false, A.v, false, B0.v,
                                                           (short)0, acc0, false, false);
            acc1 = __builtin_amdgcn_wmma_f32_16x16x32_bf16(false, A.v, false, B1.v,
                                                           (short)0, acc1, false, false);
        }
    }

    int n0 = lane & 15;
    float b0v = bias[n0], b1v = bias[n0 + 16];
    float al = *alphap;
#pragma unroll
    for (int r = 0; r < 8; ++r) {
        int mm = r + 8 * h;
        size_t pix = (size_t)y * WW + x0 + mm;
        float v0 = acc0[r] + b0v;
        v0 = (v0 >= 0.f) ? v0 : al * v0;
        float v1 = acc1[r] + b1v;
        v1 = (v1 >= 0.f) ? v1 : al * v1;
        out[pix * 32 + n0]      = f2bfbits(v0);
        out[pix * 32 + n0 + 16] = f2bfbits(v1);
    }
}

// ---------------------------------------------------------------------------
// dec3: Cin = 1 (x, f32) + 32 (x5 bf16 HWC) -> 7 logits; 6*sigmoid, round,
// emit 6 class planes (uint8).  32-channel part via WMMA (1 N-tile, 7 valid).
// ---------------------------------------------------------------------------
__global__ __launch_bounds__(256)
void dec3_cls(const float* __restrict__ x, const unsigned short* __restrict__ x5,
              const uint32_t* __restrict__ pw, const float* __restrict__ bias,
              const float* __restrict__ w33, unsigned char* __restrict__ cls) {
    int lane = threadIdx.x & 31;
    int wave = blockIdx.x * (blockDim.x >> 5) + (threadIdx.x >> 5);
    int xt = wave % (WW / 16);
    int y  = wave / (WW / 16);
    if (y >= HH) return;
    int x0 = xt * 16;
    int m = lane & 15, h = lane >> 4;

    v8f acc0 = {0.f, 0.f, 0.f, 0.f, 0.f, 0.f, 0.f, 0.f};
#pragma unroll
    for (int t = 0; t < 9; ++t) {
        int py = y + t / 3 - 1;
        int px = x0 + m + t % 3 - 1;
        V16BF A;
        if (py >= 0 && py < HH && px >= 0 && px < WW) {
            const uint4* row = (const uint4*)(x5 + ((size_t)py * WW + px) * 32);
            A.q[0] = row[2 * h];
            A.q[1] = row[2 * h + 1];
        } else {
            A.q[0] = make_uint4(0, 0, 0, 0);
            A.q[1] = make_uint4(0, 0, 0, 0);
        }
        const uint4* bp = (const uint4*)(pw + ((size_t)(t * 32 + lane)) * 8);
        V16BF B0;
        B0.q[0] = bp[0]; B0.q[1] = bp[1];
        acc0 = __builtin_amdgcn_wmma_f32_16x16x32_bf16(false, A.v, false, B0.v,
                                                       (short)0, acc0, false, false);
    }

    int n = lane & 15;
    float wq[9];
#pragma unroll
    for (int t = 0; t < 9; ++t) wq[t] = (n < 7) ? w33[n * 297 + t] : 0.f; // i=0 plane
    float bn = (n < 7) ? bias[n] : 0.f;

#pragma unroll
    for (int r = 0; r < 8; ++r) {
        int mm = r + 8 * h;
        int pxc = x0 + mm;
        float xc = 0.f;
#pragma unroll
        for (int t = 0; t < 9; ++t) {
            int py = y + t / 3 - 1, px = pxc + t % 3 - 1;
            float xv = (py >= 0 && py < HH && px >= 0 && px < WW) ? x[py * WW + px] : 0.f;
            xc += wq[t] * xv;
        }
        float v = acc0[r] + bn + xc;
        float sgm = 6.f / (1.f + __expf(-v));
        int c = (int)rintf(sgm);
        if (n < 6) cls[(size_t)n * HWPX + (size_t)y * WW + pxc] = (unsigned char)c;
    }
}

// ---------------------------------------------------------------------------
// One filtering iteration: per pixel select among 7 fixed kernels by class.
// k0: 16c - sum5 ; k1: 5c - sum3 ; k2: c ; k3..k6: box means 3/5/7/9.
// 16x16 tile + halo 4 in LDS, zero padding.
// ---------------------------------------------------------------------------
__global__ __launch_bounds__(256)
void filter_step(const float* __restrict__ cur, const unsigned char* __restrict__ cls,
                 float* __restrict__ out) {
    __shared__ float tile[24][24];
    int bx = blockIdx.x % (WW / 16);
    int by = blockIdx.x / (WW / 16);
    int x0 = bx * 16 - 4, y0 = by * 16 - 4;
    int tid = threadIdx.x;
    for (int i = tid; i < 24 * 24; i += 256) {
        int ly = i / 24, lx = i % 24;
        int gx = x0 + lx, gy = y0 + ly;
        tile[ly][lx] = (gx >= 0 && gx < WW && gy >= 0 && gy < HH) ? cur[(size_t)gy * WW + gx] : 0.f;
    }
    __syncthreads();
    int lx = tid & 15, ly = tid >> 4;
    int gx = bx * 16 + lx, gy = by * 16 + ly;
    int k = cls[(size_t)gy * WW + gx];
    int cx = lx + 4, cy = ly + 4;
    float c = tile[cy][cx];
    const int radTab[7] = {2, 1, 0, 1, 2, 3, 4};
    int rad = radTab[k > 6 ? 6 : k];
    float s = 0.f;
    for (int dy = -rad; dy <= rad; ++dy)
        for (int dx = -rad; dx <= rad; ++dx)
            s += tile[cy + dy][cx + dx];
    float o;
    switch (k) {
        case 0: o = 16.f * c - s; break;
        case 1: o = 5.f * c - s; break;
        case 2: o = c; break;
        case 3: o = s * (1.f / 9.f); break;
        case 4: o = s * (1.f / 25.f); break;
        case 5: o = s * (1.f / 49.f); break;
        default: o = s * (1.f / 81.f); break;
    }
    out[(size_t)gy * WW + gx] = o;
}

// ---------------------------------------------------------------------------
// Host orchestration.  Input order = pytree(sorted params keys) after x, y:
// 0:x 1:y 2:dec1_b 3:dec1_w 4:dec2_b 5:dec2_w 6:dec3_b 7:dec3_w 8:enc1_b
// 9:enc1_w 10:enc2_b 11:enc2_w 12:enc3_b 13:enc3_w 14..20:fixed kernels
// 21:nir_b1 22:nir_b2 23:nir_prelu 24:nir_w1 25:nir_w2 26:prelu
// ---------------------------------------------------------------------------
extern "C" void kernel_launch(void* const* d_in, const int* in_sizes, int n_in,
                              void* d_out, int out_size, void* d_ws, size_t ws_size,
                              hipStream_t stream) {
    (void)in_sizes; (void)n_in; (void)out_size; (void)ws_size;
    const float* x      = (const float*)d_in[0];
    const float* y      = (const float*)d_in[1];
    const float* dec1_b = (const float*)d_in[2];
    const float* dec1_w = (const float*)d_in[3];
    const float* dec2_b = (const float*)d_in[4];
    const float* dec2_w = (const float*)d_in[5];
    const float* dec3_b = (const float*)d_in[6];
    const float* dec3_w = (const float*)d_in[7];
    const float* enc1_b = (const float*)d_in[8];
    const float* enc1_w = (const float*)d_in[9];
    const float* enc2_b = (const float*)d_in[10];
    const float* enc2_w = (const float*)d_in[11];
    const float* enc3_b = (const float*)d_in[12];
    const float* enc3_w = (const float*)d_in[13];
    const float* nir_b1 = (const float*)d_in[21];
    const float* nir_b2 = (const float*)d_in[22];
    const float* nir_a  = (const float*)d_in[23];
    const float* nir_w1 = (const float*)d_in[24];
    const float* nir_w2 = (const float*)d_in[25];
    const float* prelu  = (const float*)d_in[26];

    uint8_t* ws = (uint8_t*)d_ws;
    uint32_t* pw_nir2 = (uint32_t*)(ws + 0);
    uint32_t* pw_enc2 = (uint32_t*)(ws + 18432);
    uint32_t* pw_enc3 = (uint32_t*)(ws + 36864);
    uint32_t* pw_dec1 = (uint32_t*)(ws + 55296);
    uint32_t* pw_dec2 = (uint32_t*)(ws + 92160);
    uint32_t* pw_dec3 = (uint32_t*)(ws + 129024);
    const size_t FEAT = (size_t)HWPX * 32 * 2; // 64 MB bf16 HWC
    unsigned short* B0 = (unsigned short*)(ws + 262144);
    unsigned short* B1 = (unsigned short*)(ws + 262144 + FEAT);
    unsigned short* B2 = (unsigned short*)(ws + 262144 + 2 * FEAT);
    unsigned short* B3 = (unsigned short*)(ws + 262144 + 3 * FEAT);
    unsigned char*  cls = (unsigned char*)(ws + 262144 + 4 * FEAT);
    float* curA = (float*)(ws + 262144 + 4 * FEAT + 6 * (size_t)HWPX);
    float* curB = (float*)((uint8_t*)curA + (size_t)HWPX * 4);

    // --- pack weights (bf16 B-matrix layout) ---
    auto prep = [&](const float* w, uint32_t* dst, int CinW, int ci0, int G, int NT, int CoutV) {
        int total = G * 9 * NT * 256;
        prep_pack<<<(total + 255) / 256, 256, 0, stream>>>(w, dst, CinW, ci0, G, NT, CoutV);
    };
    prep(nir_w2, pw_nir2, 32, 0, 1, 2, 32);
    prep(enc2_w, pw_enc2, 32, 0, 1, 2, 32);
    prep(enc3_w, pw_enc3, 32, 0, 1, 2, 32);
    prep(dec1_w, pw_dec1, 64, 0, 2, 2, 32);
    prep(dec2_w, pw_dec2, 64, 0, 2, 2, 32);
    prep(dec3_w, pw_dec3, 33, 1, 1, 1, 7);

    const int pixBlocks  = HWPX / 256;               // 4096
    const int tileBlocks = (WW / 16) * (HH / 8);     // 8192 (8x16 output tiles)
    const int waveBlocks = (HH * (WW / 16)) / 8;     // 8192 (dec3: 1 wave per row seg)

    // nir path
    conv1to32<<<pixBlocks, 256, 0, stream>>>(y, nir_w1, nir_b1, nir_a, nullptr, B0);
    conv32_lds<1><<<tileBlocks, 256, 0, stream>>>(B0, nullptr, pw_nir2, nir_b2, nir_a, B1);
    // x1 = prelu(enc1(x)) + nir
    conv1to32<<<pixBlocks, 256, 0, stream>>>(x, enc1_w, enc1_b, prelu, B1, B2);
    // x2, x3
    conv32_lds<1><<<tileBlocks, 256, 0, stream>>>(B2, nullptr, pw_enc2, enc2_b, prelu, B0);
    conv32_lds<1><<<tileBlocks, 256, 0, stream>>>(B0, nullptr, pw_enc3, enc3_b, prelu, B1);
    // x4 = dec1(cat(x2,x3)), x5 = dec2(cat(x1,x4))
    conv32_lds<2><<<tileBlocks, 256, 0, stream>>>(B0, B1, pw_dec1, dec1_b, prelu, B3);
    conv32_lds<2><<<tileBlocks, 256, 0, stream>>>(B2, B3, pw_dec2, dec2_b, prelu, B0);
    // class planes
    dec3_cls<<<waveBlocks, 256, 0, stream>>>(x, B0, pw_dec3, dec3_b, dec3_w, cls);

    // 6 select-filter iterations (cur starts at x; last writes d_out)
    filter_step<<<pixBlocks, 256, 0, stream>>>(x,    cls + 0 * (size_t)HWPX, curA);
    filter_step<<<pixBlocks, 256, 0, stream>>>(curA, cls + 1 * (size_t)HWPX, curB);
    filter_step<<<pixBlocks, 256, 0, stream>>>(curB, cls + 2 * (size_t)HWPX, curA);
    filter_step<<<pixBlocks, 256, 0, stream>>>(curA, cls + 3 * (size_t)HWPX, curB);
    filter_step<<<pixBlocks, 256, 0, stream>>>(curB, cls + 4 * (size_t)HWPX, curA);
    filter_step<<<pixBlocks, 256, 0, stream>>>(curA, cls + 5 * (size_t)HWPX, (float*)d_out);
}
